// AttentionModulator_19353122635924
// MI455X (gfx1250) — matrix operand order repaired and checked by reading, not compiled, not run on hardware
//
#include <hip/hip_runtime.h>
#include <math.h>

typedef unsigned short u16;
typedef __attribute__((ext_vector_type(16))) __bf16 v16bf;
typedef __attribute__((ext_vector_type(8)))  float  v8f;

// ---------------- model constants ----------------
constexpr int kBS   = 4;
constexpr int kN    = 1024;
constexpr int kDN   = 128;
constexpr int kDP   = 64;
constexpr int kRD   = 16;
constexpr int kKout = 512;
constexpr int kF    = 512;
constexpr int kH    = 8;
constexpr int kL    = 4;
constexpr int kHD   = 64;
constexpr int kFF   = 2048;
constexpr int kTOK  = 213;
constexpr int kTOKP = 224;          // K padded to multiple of 32
constexpr int kM    = kBS * kN;     // 4096 token rows

// ---------------- small helpers ----------------
__device__ __forceinline__ u16 f2bf(float f) {
  unsigned u = __float_as_uint(f);
  u += 0x7FFFu + ((u >> 16) & 1u);       // round-to-nearest-even
  return (u16)(u >> 16);
}
__device__ __forceinline__ float bf2f(u16 h) {
  return __uint_as_float(((unsigned)h) << 16);
}
__device__ __forceinline__ float gelu_f(float x) {
  return 0.5f * x * (1.0f + erff(x * 0.70710678118654752f));
}

__device__ __forceinline__ v16bf frag_from_u(const unsigned* w) {
  union { unsigned u[8]; v16bf v; } f;
#pragma unroll
  for (int i = 0; i < 8; i++) f.u[i] = w[i];
  return f.v;
}

// A-matrix fragment (16x32 bf16), A row-major [*, lda], lda mult of 8, k0 mult of 32.
// ISA layout: lanes 0-15 -> M=lane, K in {kb..kb+7} U {kb+16..kb+23}, kb=0; lanes 16-31: kb=8.
__device__ __forceinline__ v16bf load_frag_a(const u16* A, int lda, int row0, int k0, int lane) {
  const int m  = lane & 15;
  const int kb = (lane & 16) ? 8 : 0;
  const u16* p = A + (size_t)(row0 + m) * lda + k0 + kb;
  uint4 x0 = *reinterpret_cast<const uint4*>(p);
  uint4 x1 = *reinterpret_cast<const uint4*>(p + 16);
  unsigned u[8] = {x0.x, x0.y, x0.z, x0.w, x1.x, x1.y, x1.z, x1.w};
  return frag_from_u(u);
}

// B-matrix fragment (32x16 bf16) from B^T stored [Ncol][K] (K contiguous).
// ISA layout: lane = column N, K in {kb..kb+7} U {kb+16..kb+23}, kb=0 for lanes 0-15; lanes 16-31: kb=8.
__device__ __forceinline__ v16bf load_frag_b(const u16* Bt, int ldb, int n0, int k0, int lane) {
  const int n  = lane & 15;
  const int kb = (lane & 16) ? 8 : 0;
  const u16* p = Bt + (size_t)(n0 + n) * ldb + k0 + kb;
  uint4 x0 = *reinterpret_cast<const uint4*>(p);
  uint4 x1 = *reinterpret_cast<const uint4*>(p + 16);
  unsigned u[8] = {x0.x, x0.y, x0.z, x0.w, x1.x, x1.y, x1.z, x1.w};
  return frag_from_u(u);
}

__device__ __forceinline__ v8f wmma_bf16(v16bf a, v16bf b, v8f c) {
  return __builtin_amdgcn_wmma_f32_16x16x32_bf16(false, a, false, b, (short)0, c, false, false);
}

// ---------------- weight convert + transpose (f32 [K][N] -> bf16 [N][Kpad]) ----------------
__global__ void conv_t_kernel(const float* __restrict__ W, u16* __restrict__ Wt,
                              int Ksrc, int Ncol, int Kpad) {
  size_t gid = (size_t)blockIdx.x * 256 + threadIdx.x;
  size_t total = (size_t)Ncol * Kpad;
  if (gid >= total) return;
  int k = (int)(gid % Kpad);
  int n = (int)(gid / Kpad);
  float v = (k < Ksrc) ? W[(size_t)k * Ncol + n] : 0.0f;
  Wt[gid] = f2bf(v);
}

// ---------------- token feature builder ----------------
__global__ void build_tok_kernel(const float* __restrict__ h, const float* __restrict__ msg,
                                 const float* __restrict__ rcv, const float* __restrict__ decay,
                                 const float* __restrict__ s_live, const float* __restrict__ s_ema,
                                 const int* __restrict__ role_id, const float* __restrict__ Wh,
                                 const float* __restrict__ Wme, const float* __restrict__ Wmr,
                                 const float* __restrict__ role_emb, u16* __restrict__ tok) {
  const int row = blockIdx.x;          // 0..4095
  const int n = row & (kN - 1);
  const int bz = row >> 10;
  const int t = threadIdx.x;           // 64 threads
  const float* hr = h   + (size_t)row * kDN;
  const float* mr = msg + (size_t)row * kDN;
  const float* rr = rcv + (size_t)row * kDN;
  u16* tr = tok + (size_t)row * kTOKP;
  float a0 = 0.f, a1 = 0.f, a2 = 0.f;
#pragma unroll 4
  for (int i = 0; i < kDN; i++) {
    a0 += hr[i] * Wh [i * kDP + t];
    a1 += mr[i] * Wme[i * kDP + t];
    a2 += rr[i] * Wmr[i * kDP + t];
  }
  tr[3 + t]           = f2bf(a0);
  tr[3 + kDP + t]     = f2bf(a1);
  tr[3 + 2 * kDP + t] = f2bf(a2);
  if (t < kRD) tr[3 + 3 * kDP + t] = f2bf(role_emb[role_id[n] * kRD + t]);
  if (t == 0) { float s = 0.f; for (int i = 0; i < kDN; i++) s += hr[i] * hr[i]; tr[0] = f2bf(sqrtf(s)); }
  if (t == 1) { float s = 0.f; for (int i = 0; i < kDN; i++) s += mr[i] * mr[i]; tr[1] = f2bf(sqrtf(s)); }
  if (t == 2) tr[2] = f2bf(decay[row]);
  if (t == 3) tr[kTOK - 2] = f2bf(s_live[bz]);
  if (t == 4) tr[kTOK - 1] = f2bf(s_ema[bz]);
  if (t >= 5 && t < 5 + (kTOKP - kTOK)) tr[kTOK + (t - 5)] = 0;   // zero the K padding
}

// ---------------- edge-bias MLP over all pairs ----------------
__global__ __launch_bounds__(256)
void edge_bias_kernel(const float* __restrict__ W, const float* __restrict__ heb,
                      const float* __restrict__ W1, const float* __restrict__ b1,
                      const float* __restrict__ W2, const float* __restrict__ b2,
                      u16* __restrict__ bias) {
  size_t gid = (size_t)blockIdx.x * 256 + threadIdx.x;   // over BS*N*N
  const int m  = (int)(gid & (kN - 1));
  size_t t = gid >> 10;
  const int n  = (int)(t & (kN - 1));
  const int bz = (int)(t >> 10);
  const float w  = W[gid];
  const float hb = heb[gid];
  const float wt = W[((size_t)bz * kN + m) * kN + n];
  const float as = w - wt;
  float hid[8];
#pragma unroll
  for (int j = 0; j < 8; j++)
    hid[j] = gelu_f(w * W1[j] + hb * W1[8 + j] + as * W1[16 + j] + b1[j]);
#pragma unroll
  for (int hh = 0; hh < 8; hh++) {
    float o = b2[hh];
#pragma unroll
    for (int j = 0; j < 8; j++) o += hid[j] * W2[j * 8 + hh];
    bias[((size_t)(bz * kH + hh) * kN + n) * kN + m] = f2bf(o);
  }
}

// ---------------- generic WMMA GEMM: C = act(A @ B^T + bias) (+resid) ----------------
// A: bf16 row-major [M][K]; Bt: bf16 [Ncol][K]; block = 8 waves, 128x64 tile, wave = 16x64.
// B tile (64 cols x 32 k = 4KB) is staged into LDS once per k-step with the CDNA5
// async global->LDS path (ASYNCcnt), double-buffered; fragments then come from LDS.
constexpr int kLdsPitch = 40;   // u16 per row (32 data + 8 pad): lane bank stride 20 mod 64 -> conflict-free

template <int ACT, bool RESID>
__global__ __launch_bounds__(256, 2)
void gemm_kernel(const u16* __restrict__ A, const u16* __restrict__ Bt,
                 const float* __restrict__ bias, const float* __restrict__ resid,
                 float* __restrict__ outF, u16* __restrict__ outB,
                 int M, int Ncol, int K) {
  __shared__ __align__(16) u16 ldsB[2][64 * kLdsPitch];
  const int tid  = threadIdx.x;
  const int lane = tid & 31;
  const int wv   = tid >> 5;
  const int row0 = blockIdx.x * 128 + wv * 16;
  const int col0 = blockIdx.y * 64;

  // per-thread slice of the async B-tile copy: 16 bytes each (256 threads x 16B = 4KB)
  const int cpyCol = tid >> 2;          // 0..63
  const int cpyK   = (tid & 3) * 8;     // 0,8,16,24
  const unsigned ldst0 = (unsigned)(uintptr_t)&ldsB[0][cpyCol * kLdsPitch + cpyK];
  const unsigned ldst1 = (unsigned)(uintptr_t)&ldsB[1][cpyCol * kLdsPitch + cpyK];
  const u16* gsrc = Bt + (size_t)(col0 + cpyCol) * K + cpyK;

  // prologue: stage k-step 0 into buffer 0
  asm volatile("global_load_async_to_lds_b128 %0, %1, off"
               :: "v"(ldst0), "v"((unsigned long long)(uintptr_t)gsrc)
               : "memory");

  v8f acc[4];
#pragma unroll
  for (int i = 0; i < 4; i++) acc[i] = (v8f)0.0f;

  const int nsteps = K >> 5;
  for (int s = 0; s < nsteps; s++) {
    if (s + 1 < nsteps) {
      const unsigned dn = ((s + 1) & 1) ? ldst1 : ldst0;
      const u16* gn = gsrc + (size_t)(s + 1) * 32;
      asm volatile("global_load_async_to_lds_b128 %0, %1, off"
                   :: "v"(dn), "v"((unsigned long long)(uintptr_t)gn)
                   : "memory");
      asm volatile("s_wait_asynccnt 0x1" ::: "memory");   // current buffer done, next in flight
    } else {
      asm volatile("s_wait_asynccnt 0x0" ::: "memory");
    }
    __syncthreads();                                      // publish LDS tile to all waves
    v16bf a = load_frag_a(A, K, row0, s * 32, lane);
    const u16* bb = &ldsB[s & 1][0];
#pragma unroll
    for (int ns = 0; ns < 4; ns++) {
      v16bf b = load_frag_b(bb, kLdsPitch, ns * 16, 0, lane);
      acc[ns] = wmma_bf16(a, b, acc[ns]);
    }
    __syncthreads();                                      // all reads done before buffer reuse
  }

  const int rOff = (lane & 16) ? 8 : 0;
  const int cl   = lane & 15;
#pragma unroll
  for (int ns = 0; ns < 4; ns++) {
#pragma unroll
    for (int r = 0; r < 8; r++) {
      const int row = row0 + rOff + r;
      const int col = col0 + ns * 16 + cl;
      float v = acc[ns][r];
      if (bias) v += bias[col];
      if (ACT == 1) v = gelu_f(v);
      const size_t idx = (size_t)row * Ncol + col;
      if (RESID) v += resid[idx];
      if (outF) outF[idx] = v;
      if (outB) outB[idx] = f2bf(v);
    }
  }
}

// ---------------- layernorm (f32 in -> bf16 out) ----------------
__global__ __launch_bounds__(256)
void ln_kernel(const float* __restrict__ x, const float* __restrict__ g,
               const float* __restrict__ bta, u16* __restrict__ out) {
  __shared__ float red[16];
  const int row = blockIdx.x;
  const float* xr = x + (size_t)row * kF;
  float v0 = xr[threadIdx.x], v1 = xr[threadIdx.x + 256];
  float s = v0 + v1, s2 = v0 * v0 + v1 * v1;
#pragma unroll
  for (int off = 16; off; off >>= 1) { s += __shfl_xor(s, off, 32); s2 += __shfl_xor(s2, off, 32); }
  const int wv = threadIdx.x >> 5, lane = threadIdx.x & 31;
  if (lane == 0) { red[wv] = s; red[8 + wv] = s2; }
  __syncthreads();
  if (threadIdx.x == 0) {
    float ts = 0.f, ts2 = 0.f;
    for (int i = 0; i < 8; i++) { ts += red[i]; ts2 += red[8 + i]; }
    red[0] = ts; red[1] = ts2;
  }
  __syncthreads();
  const float mean = red[0] * (1.0f / kF);
  const float var  = red[1] * (1.0f / kF) - mean * mean;
  const float rstd = rsqrtf(var + 1e-5f);
  out[(size_t)row * kF + threadIdx.x]       = f2bf((v0 - mean) * rstd * g[threadIdx.x] + bta[threadIdx.x]);
  out[(size_t)row * kF + threadIdx.x + 256] = f2bf((v1 - mean) * rstd * g[threadIdx.x + 256] + bta[threadIdx.x + 256]);
}

// ---------------- qkv repack: [4096][3*512] -> q[b,h,n,d], k[b,h,n,d], v^T[b,h,d,n] ----------------
__global__ void pack_qkv_kernel(const u16* __restrict__ qkv, u16* __restrict__ qp,
                                u16* __restrict__ kp, u16* __restrict__ vt) {
  size_t gid = (size_t)blockIdx.x * 256 + threadIdx.x;  // over 4096*1536
  const int col = (int)(gid % (3 * kF));
  const size_t rowg = gid / (3 * kF);
  const int n  = (int)(rowg & (kN - 1));
  const int bz = (int)(rowg >> 10);
  const int comp = col >> 9, rem = col & 511, hh = rem >> 6, d = rem & 63;
  const u16 v = qkv[gid];
  const size_t ho = (size_t)(bz * kH + hh);
  if (comp == 0)      qp[(ho * kN + n) * kHD + d] = v;
  else if (comp == 1) kp[(ho * kN + n) * kHD + d] = v;
  else                vt[(ho * kHD + d) * kN + n] = v;
}

// ---------------- flash attention with additive edge bias ----------------
__global__ __launch_bounds__(256, 2)
void attn_kernel(const u16* __restrict__ qp, const u16* __restrict__ kp,
                 const u16* __restrict__ vt, const u16* __restrict__ bias,
                 u16* __restrict__ ao) {
  __shared__ __align__(16) u16 tile[8][16][32];   // per-wave P tile (16x32 bf16)
  const int lane = threadIdx.x & 31;
  const int wv   = threadIdx.x >> 5;
  const int bz = blockIdx.z, hz = blockIdx.y;
  const int r0 = blockIdx.x * 128 + wv * 16;
  const size_t ho = (size_t)(bz * kH + hz);
  const u16* qh = qp + ho * kN * kHD;
  const u16* kh = kp + ho * kN * kHD;
  const u16* vh = vt + ho * kHD * kN;
  const u16* bh = bias + ho * kN * kN;

  const v16bf aq0 = load_frag_a(qh, kHD, r0, 0, lane);
  const v16bf aq1 = load_frag_a(qh, kHD, r0, 32, lane);

  v8f o0 = (v8f)0.f, o1 = (v8f)0.f, o2 = (v8f)0.f, o3 = (v8f)0.f;
  float mr[8], lr[8];
#pragma unroll
  for (int r = 0; r < 8; r++) { mr[r] = -3.0e38f; lr[r] = 0.0f; }
  const int rOff = (lane & 16) ? 8 : 0;
  const int cl   = lane & 15;
  const float scale = 0.125f;   // 1/sqrt(64)

  for (int m0 = 0; m0 < kN; m0 += 32) {
    // S = Q @ K^T over this 32-key chunk (two 16x16 tiles, K-dim = 64 = 2 wmma steps)
    v8f c0 = (v8f)0.f, c1 = (v8f)0.f;
    c0 = wmma_bf16(aq0, load_frag_b(kh, kHD, m0,      0,  lane), c0);
    c0 = wmma_bf16(aq1, load_frag_b(kh, kHD, m0,      32, lane), c0);
    c1 = wmma_bf16(aq0, load_frag_b(kh, kHD, m0 + 16, 0,  lane), c1);
    c1 = wmma_bf16(aq1, load_frag_b(kh, kHD, m0 + 16, 32, lane), c1);

#pragma unroll
    for (int r = 0; r < 8; r++) {
      const int row = r0 + rOff + r;
      const u16* bp = bh + (size_t)row * kN + m0;
      float s0 = c0[r] * scale + bf2f(bp[cl]);
      float s1 = c1[r] * scale + bf2f(bp[16 + cl]);
      // row-max across the 16-lane half that owns this row
      float t = fmaxf(s0, s1);
      t = fmaxf(t, __shfl_xor(t, 1, 32));
      t = fmaxf(t, __shfl_xor(t, 2, 32));
      t = fmaxf(t, __shfl_xor(t, 4, 32));
      t = fmaxf(t, __shfl_xor(t, 8, 32));
      const float nm = fmaxf(mr[r], t);
      const float p0 = __expf(s0 - nm), p1 = __expf(s1 - nm);
      float rs = p0 + p1;
      rs += __shfl_xor(rs, 1, 32);
      rs += __shfl_xor(rs, 2, 32);
      rs += __shfl_xor(rs, 4, 32);
      rs += __shfl_xor(rs, 8, 32);
      const float alpha = __expf(mr[r] - nm);
      lr[r] = lr[r] * alpha + rs;
      mr[r] = nm;
      o0[r] *= alpha; o1[r] *= alpha; o2[r] *= alpha; o3[r] *= alpha;
      const int rl = rOff + r;
      tile[wv][rl][cl]      = f2bf(p0);
      tile[wv][rl][16 + cl] = f2bf(p1);
    }
    // per-wave LDS tile: DS ops are in-order within a wave; keep compiler from reordering
    asm volatile("" ::: "memory");
    const v16bf ap = load_frag_a(&tile[wv][0][0], 32, 0, 0, lane);
    asm volatile("" ::: "memory");
    // O += P @ V  (K-dim = 32 keys, 4 column subtiles of d)
    o0 = wmma_bf16(ap, load_frag_b(vh, kN, 0,  m0, lane), o0);
    o1 = wmma_bf16(ap, load_frag_b(vh, kN, 16, m0, lane), o1);
    o2 = wmma_bf16(ap, load_frag_b(vh, kN, 32, m0, lane), o2);
    o3 = wmma_bf16(ap, load_frag_b(vh, kN, 48, m0, lane), o3);
  }
#pragma unroll
  for (int r = 0; r < 8; r++) {
    const float inv = 1.0f / lr[r];
    const int row = r0 + rOff + r;
    const size_t ob = ((size_t)bz * kN + row) * kF + hz * kHD;
    ao[ob + 0  + cl] = f2bf(o0[r] * inv);
    ao[ob + 16 + cl] = f2bf(o1[r] * inv);
    ao[ob + 32 + cl] = f2bf(o2[r] * inv);
    ao[ob + 48 + cl] = f2bf(o3[r] * inv);
  }
}

// ---------------- pooling / head / misc ----------------
__global__ void zero_kernel(float* p, int n) {
  int i = blockIdx.x * 256 + threadIdx.x;
  if (i < n) p[i] = 0.0f;
}

__global__ __launch_bounds__(256)
void pool_ln_kernel(const float* __restrict__ x, const float* __restrict__ g,
                    const float* __restrict__ bta, float* __restrict__ pooled) {
  __shared__ float red[16];
  const int row = blockIdx.x;
  const int bz = row >> 10;
  const float* xr = x + (size_t)row * kF;
  float v0 = xr[threadIdx.x], v1 = xr[threadIdx.x + 256];
  float s = v0 + v1, s2 = v0 * v0 + v1 * v1;
#pragma unroll
  for (int off = 16; off; off >>= 1) { s += __shfl_xor(s, off, 32); s2 += __shfl_xor(s2, off, 32); }
  const int wv = threadIdx.x >> 5, lane = threadIdx.x & 31;
  if (lane == 0) { red[wv] = s; red[8 + wv] = s2; }
  __syncthreads();
  if (threadIdx.x == 0) {
    float ts = 0.f, ts2 = 0.f;
    for (int i = 0; i < 8; i++) { ts += red[i]; ts2 += red[8 + i]; }
    red[0] = ts; red[1] = ts2;
  }
  __syncthreads();
  const float mean = red[0] * (1.0f / kF);
  const float var  = red[1] * (1.0f / kF) - mean * mean;
  const float rstd = rsqrtf(var + 1e-5f);
  const float y0 = (v0 - mean) * rstd * g[threadIdx.x] + bta[threadIdx.x];
  const float y1 = (v1 - mean) * rstd * g[threadIdx.x + 256] + bta[threadIdx.x + 256];
  atomicAdd(&pooled[bz * kF + threadIdx.x],       y0 * (1.0f / kN));
  atomicAdd(&pooled[bz * kF + threadIdx.x + 256], y1 * (1.0f / kN));
}

__global__ void head_kernel(const float* __restrict__ pooled, const float* __restrict__ hW,
                            const float* __restrict__ hb, float* __restrict__ logits) {
  const int idx = blockIdx.x * 256 + threadIdx.x;  // BS*Kout = 2048
  const int bz = idx >> 9;
  const int k  = idx & 511;
  float s = hb[k];
  const float* p = pooled + bz * kF;
#pragma unroll 4
  for (int f = 0; f < kF; f++) s += p[f] * hW[(size_t)f * kKout + k];
  logits[idx] = s;
}

__global__ void copy_kernel(const float* __restrict__ src, float* __restrict__ dst, int n) {
  int i = blockIdx.x * 256 + threadIdx.x;
  if (i < n) dst[i] = src[i];
}

// ---------------- host orchestration ----------------
extern "C" void kernel_launch(void* const* d_in, const int* in_sizes, int n_in,
                              void* d_out, int out_size, void* d_ws, size_t ws_size,
                              hipStream_t stream) {
  (void)in_sizes; (void)n_in; (void)out_size; (void)ws_size;
  const float* h      = (const float*)d_in[0];
  const float* msg    = (const float*)d_in[1];
  const float* rcv    = (const float*)d_in[2];
  const float* W      = (const float*)d_in[3];
  const float* heb    = (const float*)d_in[4];
  const float* decay  = (const float*)d_in[5];
  const float* s_live = (const float*)d_in[6];
  const float* s_ema  = (const float*)d_in[7];
  const int*   role_id = (const int*)d_in[8];
  const float* Wh     = (const float*)d_in[9];
  const float* Wme    = (const float*)d_in[10];
  const float* Wmr    = (const float*)d_in[11];
  const float* role_emb = (const float*)d_in[12];
  const float* tok_W1 = (const float*)d_in[13];
  const float* tok_b1 = (const float*)d_in[14];
  const float* tok_W2 = (const float*)d_in[15];
  const float* tok_b2 = (const float*)d_in[16];
  const float* eb_W1  = (const float*)d_in[17];
  const float* eb_b1  = (const float*)d_in[18];
  const float* eb_W2  = (const float*)d_in[19];
  const float* eb_b2  = (const float*)d_in[20];
  const float* ln1_g  = (const float*)d_in[21];
  const float* ln1_b  = (const float*)d_in[22];
  const float* qkv_W  = (const float*)d_in[23];
  const float* out_W  = (const float*)d_in[24];
  const float* ln2_g  = (const float*)d_in[25];
  const float* ln2_b  = (const float*)d_in[26];
  const float* ffn_W1 = (const float*)d_in[27];
  const float* ffn_b1 = (const float*)d_in[28];
  const float* ffn_W2 = (const float*)d_in[29];
  const float* ffn_b2 = (const float*)d_in[30];
  const float* pool_g = (const float*)d_in[31];
  const float* pool_b = (const float*)d_in[32];
  const float* head_W = (const float*)d_in[33];
  const float* head_b = (const float*)d_in[34];

  char* wsb = (char*)d_ws;
  size_t off = 0;
  auto alloc = [&](size_t bytes) -> char* {
    char* p = wsb + off;
    off = (off + bytes + 255) & ~(size_t)255;
    return p;
  };
  u16*   tokW1t = (u16*)alloc((size_t)kF * kTOKP * 2);
  u16*   tokW2t = (u16*)alloc((size_t)kF * kF * 2);
  u16*   qkvWt  = (u16*)alloc((size_t)kL * 3 * kF * kF * 2);
  u16*   outWt  = (u16*)alloc((size_t)kL * kF * kF * 2);
  u16*   f1Wt   = (u16*)alloc((size_t)kL * kFF * kF * 2);
  u16*   f2Wt   = (u16*)alloc((size_t)kL * kF * kFF * 2);
  u16*   tokin  = (u16*)alloc((size_t)kM * kTOKP * 2);
  u16*   hid    = (u16*)alloc((size_t)kM * kF * 2);
  float* x      = (float*)alloc((size_t)kM * kF * 4);
  u16*   hn     = (u16*)alloc((size_t)kM * kF * 2);
  u16*   qkvb   = (u16*)alloc((size_t)kM * 3 * kF * 2);
  u16*   qpk    = (u16*)alloc((size_t)kBS * kH * kN * kHD * 2);
  u16*   kpk    = (u16*)alloc((size_t)kBS * kH * kN * kHD * 2);
  u16*   vtb    = (u16*)alloc((size_t)kBS * kH * kHD * kN * 2);
  u16*   aob    = (u16*)alloc((size_t)kM * kF * 2);
  u16*   ffh    = (u16*)alloc((size_t)kM * kFF * 2);
  u16*   ebias  = (u16*)alloc((size_t)kBS * kH * kN * kN * 2);
  float* pooled = (float*)alloc((size_t)kBS * kF * 4);

  const dim3 blk(256);

  // ---- weight conversion (bf16, transposed to [N][K]) ----
  conv_t_kernel<<<(kF * kTOKP + 255) / 256, blk, 0, stream>>>(tok_W1, tokW1t, kTOK, kF, kTOKP);
  conv_t_kernel<<<(kF * kF + 255) / 256, blk, 0, stream>>>(tok_W2, tokW2t, kF, kF, kF);
  for (int l = 0; l < kL; l++) {
    conv_t_kernel<<<(3 * kF * kF + 255) / 256, blk, 0, stream>>>(
        qkv_W + (size_t)l * kF * 3 * kF, qkvWt + (size_t)l * 3 * kF * kF, kF, 3 * kF, kF);
    conv_t_kernel<<<(kF * kF + 255) / 256, blk, 0, stream>>>(
        out_W + (size_t)l * kF * kF, outWt + (size_t)l * kF * kF, kF, kF, kF);
    conv_t_kernel<<<(kFF * kF + 255) / 256, blk, 0, stream>>>(
        ffn_W1 + (size_t)l * kF * kFF, f1Wt + (size_t)l * kFF * kF, kF, kFF, kF);
    conv_t_kernel<<<(kF * kFF + 255) / 256, blk, 0, stream>>>(
        ffn_W2 + (size_t)l * kFF * kF, f2Wt + (size_t)l * kF * kFF, kFF, kF, kFF);
  }

  // ---- token features + tokenizer MLP (WMMA) ----
  build_tok_kernel<<<kM, dim3(64), 0, stream>>>(h, msg, rcv, decay, s_live, s_ema, role_id,
                                                Wh, Wme, Wmr, role_emb, tokin);
  gemm_kernel<1, false><<<dim3(kM / 128, kF / 64), blk, 0, stream>>>(
      tokin, tokW1t, tok_b1, nullptr, nullptr, hid, kM, kF, kTOKP);
  gemm_kernel<0, false><<<dim3(kM / 128, kF / 64), blk, 0, stream>>>(
      hid, tokW2t, tok_b2, nullptr, x, nullptr, kM, kF, kF);

  // ---- edge bias (computed once, reused by all layers) ----
  edge_bias_kernel<<<(kBS * kN * kN) / 256, blk, 0, stream>>>(W, heb, eb_W1, eb_b1, eb_W2, eb_b2, ebias);

  // ---- transformer layers ----
  for (int l = 0; l < kL; l++) {
    ln_kernel<<<kM, blk, 0, stream>>>(x, ln1_g + l * kF, ln1_b + l * kF, hn);
    gemm_kernel<0, false><<<dim3(kM / 128, (3 * kF) / 64), blk, 0, stream>>>(
        hn, qkvWt + (size_t)l * 3 * kF * kF, nullptr, nullptr, nullptr, qkvb, kM, 3 * kF, kF);
    pack_qkv_kernel<<<((size_t)kM * 3 * kF) / 256, blk, 0, stream>>>(qkvb, qpk, kpk, vtb);
    attn_kernel<<<dim3(kN / 128, kH, kBS), blk, 0, stream>>>(qpk, kpk, vtb, ebias, aob);
    gemm_kernel<0, true><<<dim3(kM / 128, kF / 64), blk, 0, stream>>>(
        aob, outWt + (size_t)l * kF * kF, nullptr, x, x, nullptr, kM, kF, kF);
    ln_kernel<<<kM, blk, 0, stream>>>(x, ln2_g + l * kF, ln2_b + l * kF, hn);
    gemm_kernel<1, false><<<dim3(kM / 128, kFF / 64), blk, 0, stream>>>(
        hn, f1Wt + (size_t)l * kFF * kF, ffn_b1 + l * kFF, nullptr, nullptr, ffh, kM, kFF, kF);
    gemm_kernel<0, true><<<dim3(kM / 128, kF / 64), blk, 0, stream>>>(
        ffh, f2Wt + (size_t)l * kF * kFF, ffn_b2 + l * kF, x, x, nullptr, kM, kF, kFF);
  }

  // ---- pool + head + output ----
  zero_kernel<<<(kBS * kF + 255) / 256, blk, 0, stream>>>(pooled, kBS * kF);
  pool_ln_kernel<<<kM, blk, 0, stream>>>(x, pool_g, pool_b, pooled);
  head_kernel<<<(kBS * kKout + 255) / 256, blk, 0, stream>>>(pooled, head_W, head_b, (float*)d_out);
  copy_kernel<<<(kM * kF + 255) / 256, blk, 0, stream>>>(x, (float*)d_out + kBS * kKout, kM * kF);
}
